// RPN_45672682226161
// MI455X (gfx1250) — matrix operand order repaired and checked by reading, not compiled, not run on hardware
//
#include <hip/hip_runtime.h>
#include <stdint.h>

typedef __attribute__((ext_vector_type(16))) _Float16 v16h;
typedef __attribute__((ext_vector_type(8)))  float    v8f;
typedef _Float16 f16;

#define NB      16
#define CIN     256
#define KTOT    2304          // 9 * 256
#define NPROP   1000
#define NTOT    64512         // 3*(128*128 + 64*64 + 32*32)
#define NKEYS   65536
#define IOU_T   0.7f
#define SCALE_CLAMP 4.135166556742356f   // log(1000/16)

// ---------------- packing kernels ----------------

// zero-fill the 256-halfword padding page used for out-of-bounds conv taps
__global__ void zero_pad_kernel(f16* __restrict__ zpad) {
  int i = threadIdx.x;   // 256 threads
  zpad[i] = (f16)0.f;
}

// NCHW f32 -> NHWC f16 (channels innermost)
__global__ void pack_nhwc_kernel(const float* __restrict__ src, f16* __restrict__ dst,
                                 int H, int W) {
  size_t i = (size_t)blockIdx.x * blockDim.x + threadIdx.x;
  size_t total = (size_t)NB * CIN * H * W;
  if (i >= total) return;
  int co = (int)(i & (CIN - 1));
  size_t p = i >> 8;
  int x = (int)(p % W);
  int y = (int)((p / W) % H);
  int n = (int)(p / ((size_t)W * H));
  dst[i] = (f16)src[(((size_t)n * CIN + co) * H + y) * W + x];
}

// OIHW f32 -> [cout_pad][k] f16 with k = tap*256 + ci ; pad channels are zero
__global__ void pack_w_kernel(const float* __restrict__ w, f16* __restrict__ dst,
                              int cout_valid, int cout_pad) {
  int i = blockIdx.x * blockDim.x + threadIdx.x;
  int total = cout_pad * KTOT;
  if (i >= total) return;
  int k  = i % KTOT;
  int co = i / KTOT;
  int tap = k >> 8;
  int ci  = k & 255;
  f16 v = (f16)0.f;
  if (co < cout_valid) v = (f16)w[((size_t)co * CIN + ci) * 9 + tap];
  dst[i] = v;
}

// ---------------- implicit-GEMM WMMA convolutions ----------------
// Fragment layouts (16-bit, wave32):
//   A 16x32: lanes 0-15 hold M=lane, K in {kb..kb+7, kb+16..kb+23}, kb = 8*(lane>=16)
//   B 32x16: lane holds N=lane&15, K in {kb..kb+7, kb+16..kb+23}, kb = 8*(lane>=16)
//   C/D 16x16: N = lane&15, M = vgpr + 8*(lane>=16)
// Out-of-bounds taps read a zero page so fragment loads are unconditional
// (no EXEC churn; WMMA requires EXEC all-ones anyway).

// conv1: 32-pixel x 32-outchannel tile per wave (2x2 WMMA register blocking)
// -> each A/B fragment is reused by two WMMAs: 2 x b128 loads per WMMA.
__global__ void conv1_wmma_kernel(const f16* __restrict__ act, const f16* __restrict__ wpack,
                                  const float* __restrict__ bias, const f16* __restrict__ zpad,
                                  f16* __restrict__ out, int H, int W) {
  int lane = threadIdx.x & 31;
  int gw = blockIdx.x * (blockDim.x >> 5) + (threadIdx.x >> 5);
  int coT = gw & 7;                 // 8 tiles of 32 output channels
  int pt  = gw >> 3;
  int WT  = W >> 5;                 // 32-pixel tiles along x
  int x0  = (pt % WT) << 5;
  int t2  = pt / WT;
  int y   = t2 % H;
  int n   = t2 / H;
  int co0 = coT << 5;

  int nlo    = lane & 15;
  int half   = lane >> 4;
  int klaneA = half << 3;     // 0 or 8
  int kbB    = half << 3;     // 0 or 8
  int xA0 = x0 + nlo;         // A row (pixel) for M-half 0
  int xA1 = xA0 + 16;         // A row (pixel) for M-half 1

  v8f c00 = {}, c01 = {}, c10 = {}, c11 = {};
  const f16* wb0 = wpack + (size_t)(co0 + nlo) * KTOT + kbB;        // N-half 0
  const f16* wb1 = wpack + (size_t)(co0 + 16 + nlo) * KTOT + kbB;   // N-half 1
  const f16* zp  = zpad + klaneA;

  for (int tap = 0; tap < 9; ++tap) {
    int ky = tap / 3, kx = tap - ky * 3;
    int iy = y + ky - 1;
    int ix0 = xA0 + kx - 1;
    int ix1 = xA1 + kx - 1;
    bool okY = ((unsigned)iy < (unsigned)H);
    bool v0 = okY && ((unsigned)ix0 < (unsigned)W);
    bool v1 = okY && ((unsigned)ix1 < (unsigned)W);
    size_t rowb = (((size_t)n * H + iy) * W) << 8;
    const f16* ap0 = v0 ? act + rowb + (((size_t)ix0) << 8) + klaneA : zp;
    const f16* ap1 = v1 ? act + rowb + (((size_t)ix1) << 8) + klaneA : zp;
    const f16* bp0 = wb0 + tap * 256;
    const f16* bp1 = wb1 + tap * 256;

    #pragma unroll
    for (int cc = 0; cc < 8; ++cc) {
      v16h a0, a1, b0, b1;
      #pragma unroll
      for (int h = 0; h < 8; ++h) {
        a0[h]     = ap0[cc * 32 + h];
        a0[8 + h] = ap0[cc * 32 + 16 + h];
        a1[h]     = ap1[cc * 32 + h];
        a1[8 + h] = ap1[cc * 32 + 16 + h];
      }
      #pragma unroll
      for (int h = 0; h < 16; ++h) {
        b0[h] = bp0[cc * 32 + h + (h & 8)];
        b1[h] = bp1[cc * 32 + h + (h & 8)];
      }
      c00 = __builtin_amdgcn_wmma_f32_16x16x32_f16(false, a0, false, b0, (short)0, c00, false, false);
      c01 = __builtin_amdgcn_wmma_f32_16x16x32_f16(false, a0, false, b1, (short)0, c01, false, false);
      c10 = __builtin_amdgcn_wmma_f32_16x16x32_f16(false, a1, false, b0, (short)0, c10, false, false);
      c11 = __builtin_amdgcn_wmma_f32_16x16x32_f16(false, a1, false, b1, (short)0, c11, false, false);
    }
  }

  float bb0 = bias[co0 + nlo];
  float bb1 = bias[co0 + 16 + nlo];
  int mbase = half << 3;
  size_t rowo = (((size_t)n * H + y) * W) << 8;
  #pragma unroll
  for (int r = 0; r < 8; ++r) {
    int xo0 = x0 + mbase + r;
    int xo1 = xo0 + 16;
    float u00 = c00[r] + bb0; u00 = u00 > 0.f ? u00 : 0.f;
    float u01 = c01[r] + bb1; u01 = u01 > 0.f ? u01 : 0.f;
    float u10 = c10[r] + bb0; u10 = u10 > 0.f ? u10 : 0.f;
    float u11 = c11[r] + bb1; u11 = u11 > 0.f ? u11 : 0.f;
    out[rowo + (((size_t)xo0) << 8) + co0 + nlo]      = (f16)u00;
    out[rowo + (((size_t)xo0) << 8) + co0 + 16 + nlo] = (f16)u01;
    out[rowo + (((size_t)xo1) << 8) + co0 + nlo]      = (f16)u10;
    out[rowo + (((size_t)xo1) << 8) + co0 + 16 + nlo] = (f16)u11;
  }
}

// conv2: 16-pixel x 16-channel tile (only 15 real output channels, ~6% of FLOPs)
__global__ void conv2_wmma_kernel(const f16* __restrict__ act, const f16* __restrict__ wpack,
                                  const float* __restrict__ bias, const f16* __restrict__ zpad,
                                  float* __restrict__ out, int H, int W) {
  int lane = threadIdx.x & 31;
  int gw = blockIdx.x * (blockDim.x >> 5) + (threadIdx.x >> 5);
  int WT  = W >> 4;
  int x0  = (gw % WT) << 4;
  int t2  = gw / WT;
  int y   = t2 % H;
  int n   = t2 / H;

  int nlo    = lane & 15;
  int half   = lane >> 4;
  int klaneA = half << 3;
  int kbB    = half << 3;
  int xA = x0 + nlo;
  int co = nlo;

  v8f c = {};
  const f16* wbase = wpack + (size_t)co * KTOT + kbB;
  const f16* zp    = zpad + klaneA;

  for (int tap = 0; tap < 9; ++tap) {
    int ky = tap / 3, kx = tap - ky * 3;
    int iy = y + ky - 1, ix = xA + kx - 1;
    bool valid = ((unsigned)iy < (unsigned)H) && ((unsigned)ix < (unsigned)W);
    const f16* ap = valid
        ? act + ((((size_t)n * H + iy) * W + ix) << 8) + klaneA
        : zp;
    const f16* bp = wbase + tap * 256;

    #pragma unroll
    for (int cc = 0; cc < 8; ++cc) {
      v16h a, bfr;
      #pragma unroll
      for (int h = 0; h < 8; ++h) a[h] = ap[cc * 32 + h];
      #pragma unroll
      for (int h = 0; h < 8; ++h) a[8 + h] = ap[cc * 32 + 16 + h];
      #pragma unroll
      for (int h = 0; h < 16; ++h) bfr[h] = bp[cc * 32 + h + (h & 8)];
      c = __builtin_amdgcn_wmma_f32_16x16x32_f16(false, a, false, bfr, (short)0, c, false, false);
    }
  }

  float bb = (co < 15) ? bias[co] : 0.f;
  int mbase = half << 3;
  #pragma unroll
  for (int r = 0; r < 8; ++r) {
    int xo = x0 + mbase + r;
    size_t pix = ((size_t)n * H + y) * W + xo;
    out[(pix << 4) + co] = c[r] + bb;
  }
}

// ---------------- merge: deltas -> boxes, sigmoid -> scores ----------------
__global__ void merge_kernel(const float* __restrict__ c2out, const float* __restrict__ anchors,
                             float* __restrict__ boxes, float* __restrict__ scores,
                             int H, int W, int lvl_off) {
  int i = blockIdx.x * blockDim.x + threadIdx.x;
  int total = NB * H * W * 3;
  if (i >= total) return;
  int a = i % 3;
  int p = i / 3;
  int x = p % W;
  int y = (p / W) % H;
  int n = p / (W * H);
  size_t pix = ((size_t)n * H + y) * W + x;
  const float* d = c2out + (pix << 4) + a * 5;
  float dx = d[0], dy = d[1];
  float dw = fminf(d[2], SCALE_CLAMP);
  float dh = fminf(d[3], SCALE_CLAMP);
  float logit = d[4];
  int arow = (y * W + x) * 3 + a;
  const float* an = anchors + (size_t)arow * 4;
  float aw = an[2] - an[0], ah = an[3] - an[1];
  float cx = an[0] + 0.5f * aw, cy = an[1] + 0.5f * ah;
  float pcx = dx * aw + cx, pcy = dy * ah + cy;
  float pw = expf(dw) * aw, ph = expf(dh) * ah;
  int g = lvl_off + arow;
  float* bo = boxes + ((size_t)n * NTOT + g) * 4;
  bo[0] = pcx - 0.5f * pw;
  bo[1] = pcy - 0.5f * ph;
  bo[2] = pcx + 0.5f * pw;
  bo[3] = pcy + 0.5f * ph;
  scores[(size_t)n * NTOT + g] = 1.f / (1.f + expf(-logit));
}

// ---------------- top-k via bitonic sort of packed keys ----------------
__global__ void build_keys_kernel(const float* __restrict__ scores, unsigned long long* __restrict__ keys) {
  int gid = blockIdx.x * blockDim.x + threadIdx.x;   // NB * NKEYS threads
  int b = gid >> 16;
  int t = gid & (NKEYS - 1);
  unsigned long long k;
  if (t < NTOT) {
    unsigned sb = __float_as_uint(scores[(size_t)b * NTOT + t]);  // scores in (0,1): bit order == value order
    k = ((unsigned long long)(~sb) << 32) | (unsigned)t;          // ascending key == descending score, ties by idx
  } else {
    k = 0xFFFFFFFFFFFFFFFFull;
  }
  keys[((size_t)b << 16) + t] = k;
}

__global__ void bitonic_step_kernel(unsigned long long* __restrict__ keys, unsigned k, unsigned j) {
  unsigned gid = blockIdx.x * blockDim.x + threadIdx.x;
  unsigned b = gid >> 16;
  unsigned t = gid & (NKEYS - 1);
  unsigned p = t ^ j;
  if (p > t) {
    unsigned long long* base = keys + ((size_t)b << 16);
    unsigned long long u = base[t], v = base[p];
    bool asc = ((t & k) == 0);
    if (asc ? (u > v) : (u < v)) { base[t] = v; base[p] = u; }
  }
}

__global__ void gather_topk_kernel(const unsigned long long* __restrict__ keys,
                                   const float* __restrict__ boxes, const float* __restrict__ scores,
                                   float* __restrict__ tb, float* __restrict__ ts, float* __restrict__ ti) {
  int gid = blockIdx.x * blockDim.x + threadIdx.x;
  if (gid >= NB * NPROP) return;
  int b = gid / NPROP;
  int p = gid - b * NPROP;
  unsigned long long key = keys[((size_t)b << 16) + p];
  unsigned idx = (unsigned)(key & 0xFFFFFFFFu);
  const float* src = boxes + ((size_t)b * NTOT + idx) * 4;
  float* dst = tb + (size_t)gid * 4;
  dst[0] = src[0]; dst[1] = src[1]; dst[2] = src[2]; dst[3] = src[3];
  ts[gid] = scores[(size_t)b * NTOT + idx];
  int lvl = (idx < 49152u) ? 0 : (idx < 61440u ? 1 : 2);
  int a = (int)(idx % 3u);                 // level offsets are multiples of 3
  ti[gid] = (float)(lvl * 3 + a);
}

// ---------------- NMS (one block per image) ----------------
__global__ void nms_kernel(const float* __restrict__ tb, int* __restrict__ keepg) {
  __shared__ float sx1[NPROP], sy1[NPROP], sx2[NPROP], sy2[NPROP], sar[NPROP];
  __shared__ int skeep[NPROP];
  int b = blockIdx.x;
  for (int i = threadIdx.x; i < NPROP; i += blockDim.x) {
    const float* bb = tb + ((size_t)b * NPROP + i) * 4;
    float x1 = bb[0], y1 = bb[1], x2 = bb[2], y2 = bb[3];
    sx1[i] = x1; sy1[i] = y1; sx2[i] = x2; sy2[i] = y2;
    sar[i] = (x2 - x1) * (y2 - y1);
    skeep[i] = 1;
  }
  __syncthreads();
  for (int i = 0; i < NPROP - 1; ++i) {
    if (skeep[i]) {
      float ix1 = sx1[i], iy1 = sy1[i], ix2 = sx2[i], iy2 = sy2[i], iar = sar[i];
      for (int j = i + 1 + (int)threadIdx.x; j < NPROP; j += (int)blockDim.x) {
        float lx = fmaxf(ix1, sx1[j]);
        float ly = fmaxf(iy1, sy1[j]);
        float rx = fminf(ix2, sx2[j]);
        float ry = fminf(iy2, sy2[j]);
        float w = fmaxf(rx - lx, 0.f);
        float h = fmaxf(ry - ly, 0.f);
        float inter = w * h;
        float iou = inter / (iar + sar[j] - inter + 1e-9f);
        if (iou > IOU_T) skeep[j] = 0;
      }
    }
    __syncthreads();
  }
  for (int i = threadIdx.x; i < NPROP; i += blockDim.x)
    keepg[b * NPROP + i] = skeep[i];
}

// ---------------- finalize: stable kept-first reorder, zero tail ----------------
__global__ void finalize_kernel(const float* __restrict__ tb, const float* __restrict__ ts,
                                const float* __restrict__ ti, const int* __restrict__ keepg,
                                float* __restrict__ out) {
  __shared__ int sc[1024];
  int b = blockIdx.x;
  int t = threadIdx.x;
  int kv = (t < NPROP) ? keepg[b * NPROP + t] : 0;
  sc[t] = kv;
  __syncthreads();
  for (int off = 1; off < 1024; off <<= 1) {
    int v = (t >= off) ? sc[t - off] : 0;
    __syncthreads();
    sc[t] += v;
    __syncthreads();
  }
  int nkeep = sc[1023];
  if (t < NPROP) {
    int excl = sc[t] - kv;                    // kept items before t
    int pos = kv ? excl : (nkeep + (t - excl));
    float* sb = out;
    float* ss = out + NB * NPROP * 4;
    float* si = out + NB * NPROP * 5;
    size_t r = (size_t)b * NPROP + pos;
    if (kv) {
      const float* src = tb + ((size_t)b * NPROP + t) * 4;
      sb[r * 4 + 0] = src[0]; sb[r * 4 + 1] = src[1];
      sb[r * 4 + 2] = src[2]; sb[r * 4 + 3] = src[3];
      ss[r] = ts[b * NPROP + t];
      si[r] = ti[b * NPROP + t];
    } else {
      sb[r * 4 + 0] = 0.f; sb[r * 4 + 1] = 0.f;
      sb[r * 4 + 2] = 0.f; sb[r * 4 + 3] = 0.f;
      ss[r] = 0.f;
      si[r] = 0.f;
    }
  }
}

// ---------------- host orchestration ----------------
extern "C" void kernel_launch(void* const* d_in, const int* in_sizes, int n_in,
                              void* d_out, int out_size, void* d_ws, size_t ws_size,
                              hipStream_t stream) {
  (void)in_sizes; (void)n_in; (void)out_size; (void)ws_size;
  // setup_inputs dict order: f0,w1_0,b1_0,w2_0,b2_0, f1,..., f2,..., anchors0,1,2
  const float* f[3]   = { (const float*)d_in[0],  (const float*)d_in[5],  (const float*)d_in[10] };
  const float* w1[3]  = { (const float*)d_in[1],  (const float*)d_in[6],  (const float*)d_in[11] };
  const float* b1[3]  = { (const float*)d_in[2],  (const float*)d_in[7],  (const float*)d_in[12] };
  const float* w2[3]  = { (const float*)d_in[3],  (const float*)d_in[8],  (const float*)d_in[13] };
  const float* b2[3]  = { (const float*)d_in[4],  (const float*)d_in[9],  (const float*)d_in[14] };
  const float* anc[3] = { (const float*)d_in[15], (const float*)d_in[16], (const float*)d_in[17] };
  const int Hs[3] = {128, 64, 32};
  const int lvl_off[3] = {0, 49152, 61440};

  size_t off = 0;
  auto take = [&](size_t bytes) -> void* {
    void* r = (char*)d_ws + off;
    off += (bytes + 255) & ~(size_t)255;
    return r;
  };
  f16* nhwc = (f16*)take((size_t)NB * 128 * 128 * CIN * 2);   // reused per level
  f16* hid  = (f16*)take((size_t)NB * 128 * 128 * CIN * 2);   // reused per level
  f16* wp1  = (f16*)take((size_t)CIN * KTOT * 2);
  f16* wp2  = (f16*)take((size_t)16 * KTOT * 2);
  f16* zpad = (f16*)take((size_t)256 * 2);                    // zero page for OOB taps
  float* c2o = (float*)take((size_t)NB * 128 * 128 * 16 * 4); // reused per level
  unsigned long long* keys = (unsigned long long*)take((size_t)NB * NKEYS * 8);
  float* tb = (float*)take((size_t)NB * NPROP * 4 * 4);
  float* ts = (float*)take((size_t)NB * NPROP * 4);
  float* ti = (float*)take((size_t)NB * NPROP * 4);
  int* keepg = (int*)take((size_t)NB * NPROP * 4);

  float* boxes_out  = (float*)d_out + NB * NPROP * 6;                 // 96000
  float* scores_out = boxes_out + (size_t)NB * NTOT * 4;              // 96000 + 4128768

  zero_pad_kernel<<<1, 256, 0, stream>>>(zpad);

  for (int lvl = 0; lvl < 3; ++lvl) {
    int H = Hs[lvl], W = Hs[lvl];
    size_t pk_total = (size_t)NB * CIN * H * W;
    pack_nhwc_kernel<<<(unsigned)((pk_total + 255) / 256), 256, 0, stream>>>(f[lvl], nhwc, H, W);
    pack_w_kernel<<<(CIN * KTOT + 255) / 256, 256, 0, stream>>>(w1[lvl], wp1, CIN, CIN);
    pack_w_kernel<<<(16 * KTOT + 255) / 256, 256, 0, stream>>>(w2[lvl], wp2, 15, 16);

    int waves1 = NB * H * (W >> 5) * 8;         // 32x32 tile per wave, 8 cout tiles
    conv1_wmma_kernel<<<waves1 / 8, 256, 0, stream>>>(nhwc, wp1, b1[lvl], zpad, hid, H, W);

    int waves2 = NB * H * (W >> 4);             // 16x16 tile (15 channels padded to 16)
    conv2_wmma_kernel<<<waves2 / 8, 256, 0, stream>>>(hid, wp2, b2[lvl], zpad, c2o, H, W);

    int mtotal = NB * H * W * 3;
    merge_kernel<<<(mtotal + 255) / 256, 256, 0, stream>>>(c2o, anc[lvl], boxes_out, scores_out,
                                                           H, W, lvl_off[lvl]);
  }

  build_keys_kernel<<<(NB * NKEYS) / 256, 256, 0, stream>>>(scores_out, keys);
  for (unsigned k = 2; k <= (unsigned)NKEYS; k <<= 1)
    for (unsigned j = k >> 1; j > 0; j >>= 1)
      bitonic_step_kernel<<<(NB * NKEYS) / 256, 256, 0, stream>>>(keys, k, j);

  gather_topk_kernel<<<(NB * NPROP + 255) / 256, 256, 0, stream>>>(keys, boxes_out, scores_out, tb, ts, ti);
  nms_kernel<<<NB, 256, 0, stream>>>(tb, keepg);
  finalize_kernel<<<NB, 1024, 0, stream>>>(tb, ts, ti, keepg, (float*)d_out);
}
